// ContrastiveLoss_V3_80410377716008
// MI455X (gfx1250) — matrix operand order repaired and checked by reading, not compiled, not run on hardware
//
#include <hip/hip_runtime.h>
#include <stdint.h>

// ---------------------------------------------------------------------------
// Contrastive loss, MI455X (gfx1250, wave32).
//
// Pass 1: per-row stats  inv = 1/max(||x||,eps), se = inv*sum(x), q = inv^2*ssq
// Pass 2: per 16-anchor tile, stage anchor/pos/neg rows to LDS via async DMA
//         (global_load_async_to_lds_b128, ASYNCcnt), compute raw dots
//         <x_i, x_pos_i>, <x_i, x_neg_i> as the DIAGONAL of a 16x16 WMMA GEMM
//         (v_wmma_f32_16x16x32_f16, K split across 4 waves), then reconstruct
//         the reference's exact distance algebra:
//           ||e_a - e_b + eps||^2 = q_a + q_b - 2*inv_a*inv_b*dot
//                                   + 2*eps*(se_a - se_b) + D*eps^2
// ---------------------------------------------------------------------------

#define N_SAMPLES 16384
#define DIM       1024
#define ROWP      (DIM + 8)          // LDS row pad: 32B shift per row vs 64 banks
#define EPS       1e-6f
#define MARGIN    1.0f

typedef __attribute__((ext_vector_type(16))) _Float16 v16h;
typedef __attribute__((ext_vector_type(8)))  float    v8f;

// async-LDS DMA builtin operand types (from hipcc diagnostic: the builtin takes
// pointers to 128-bit int vectors; src in global AS1, dst in LDS AS3)
typedef int v4i __attribute__((__vector_size__(16)));
typedef v4i __attribute__((address_space(1))) gv4i;
typedef v4i __attribute__((address_space(3))) lv4i;

__global__ void cl_zero_out(float* out) { out[0] = 0.0f; }

// ---- Pass 1: one wave per row --------------------------------------------
__launch_bounds__(256)
__global__ void cl_row_stats(const float* __restrict__ emb,
                             float* __restrict__ inv,
                             float* __restrict__ se,
                             float* __restrict__ q) {
  const int row  = blockIdx.x * 8 + (threadIdx.x >> 5);
  const int lane = threadIdx.x & 31;
  const float4* r = (const float4*)(emb + (size_t)row * DIM);
  float ssq = 0.0f, rs = 0.0f;
#pragma unroll
  for (int i = 0; i < 8; ++i) {                 // 8 * 32 lanes * float4 = 1024
    float4 v = r[lane + i * 32];
    ssq += v.x * v.x + v.y * v.y + v.z * v.z + v.w * v.w;
    rs  += v.x + v.y + v.z + v.w;
  }
#pragma unroll
  for (int off = 16; off > 0; off >>= 1) {      // wave32 reduction
    ssq += __shfl_down(ssq, off, 32);
    rs  += __shfl_down(rs,  off, 32);
  }
  if (lane == 0) {
    float nrm = sqrtf(ssq);
    float iv  = 1.0f / fmaxf(nrm, EPS);
    inv[row] = iv;
    se[row]  = iv * rs;
    q[row]   = iv * iv * ssq;
  }
}

// ---- Pass 2: 16 anchors per block, 128 threads (4 waves) ------------------
__launch_bounds__(128)
__global__ void cl_pair_loss(const float* __restrict__ emb,
                             const int* __restrict__ pidx,
                             const int* __restrict__ nidx,
                             const float* __restrict__ inv,
                             const float* __restrict__ se,
                             const float* __restrict__ q,
                             float* __restrict__ out) {
  extern __shared__ float smemf[];
  const int t    = threadIdx.x;
  const int wave = t >> 5;
  const int lane = t & 31;
  const int base = blockIdx.x * 16;

  // -------- stage 48 rows (16 anchor + 16 pos + 16 neg) into LDS ----------
  // Each thread DMAs two 16B chunks per row; all 96 transfers in flight on
  // ASYNCcnt, one wait at the end (deep pipelining, no VGPR data staging).
#pragma unroll 4
  for (int r = 0; r < 48; ++r) {
    int row;
    if (r < 16)      row = base + r;
    else if (r < 32) row = pidx[base + r - 16];
    else             row = nidx[base + r - 32];
    const float* src = emb + (size_t)row * DIM + t * 4;
    float*       dst = smemf + r * ROWP + t * 4;
#if __has_builtin(__builtin_amdgcn_global_load_async_to_lds_b128)
    __builtin_amdgcn_global_load_async_to_lds_b128(
        (gv4i*)(uintptr_t)src,
        (lv4i*)(uint32_t)(uintptr_t)dst, 0, 0);
    __builtin_amdgcn_global_load_async_to_lds_b128(
        (gv4i*)(uintptr_t)(src + 512),
        (lv4i*)(uint32_t)(uintptr_t)(dst + 512), 0, 0);
#else
    *(float4*)dst         = *(const float4*)src;
    *(float4*)(dst + 512) = *(const float4*)(src + 512);
#endif
  }
#if __has_builtin(__builtin_amdgcn_s_wait_asynccnt)
  __builtin_amdgcn_s_wait_asynccnt(0);
#endif
  __syncthreads();

  const float* sA = smemf;                // anchors [16][ROWP]
  const float* sP = smemf + 16 * ROWP;    // pos rows
  const float* sN = smemf + 32 * ROWP;    // neg rows
  float*       sC = smemf + 48 * ROWP;    // wave scratch [4][2][8][32]

  // -------- WMMA: C = A * B^T, we keep only the diagonal -------------------
  // K range split: wave w handles K = [w*256, w*256+256), 8 chunks of 32.
  // A-fragment (16x32 f16, ISA 7.12.2): lane m=l&15; K-runs {hi*8..+7} and
  // {16+hi*8..+7} with hi = l>=16. B-fragment (32x16): lane n=l&15 holds the
  // same K-runs as A: {hi*8..+7} and {16+hi*8..+7}.
  v8f cpos = {};
  v8f cneg = {};
  const int m     = lane & 15;
  const int hiA   = (lane >> 4) * 8;
  const int koffB = (lane >> 4) * 16;
  const float* arow = sA + m * ROWP;
  const float* prow = sP + m * ROWP;
  const float* nrow = sN + m * ROWP;

  for (int s = 0; s < 8; ++s) {
    const int k0 = wave * 256 + s * 32;
    v16h a, bp, bn;
#pragma unroll
    for (int j = 0; j < 8; ++j) {
      a[j]     = (_Float16)arow[k0 + hiA + j];
      a[8 + j] = (_Float16)arow[k0 + 16 + hiA + j];
    }
#pragma unroll
    for (int j = 0; j < 16; ++j) {
      bp[j] = (_Float16)prow[k0 + hiA + (j & 7) + ((j >> 3) << 4)];
      bn[j] = (_Float16)nrow[k0 + hiA + (j & 7) + ((j >> 3) << 4)];
    }
    cpos = __builtin_amdgcn_wmma_f32_16x16x32_f16(false, a, false, bp,
                                                  (short)0, cpos, false, false);
    cneg = __builtin_amdgcn_wmma_f32_16x16x32_f16(false, a, false, bn,
                                                  (short)0, cneg, false, false);
  }

  // spill accumulators so diag can be gathered across waves
#pragma unroll
  for (int r = 0; r < 8; ++r) {
    sC[((wave * 2 + 0) * 8 + r) * 32 + lane] = cpos[r];
    sC[((wave * 2 + 1) * 8 + r) * 32 + lane] = cneg[r];
  }
  __syncthreads();

  // -------- diagonal extraction + loss -------------------------------------
  // C/D layout: vgpr r, lanes 0-15 -> (M=r, N=lane); lanes 16-31 -> (M=8+r,
  // N=lane-16).  diag(m): m<8 -> vgpr m lane m; m>=8 -> vgpr m-8 lane m+16.
  if (t < 16) {
    const int vg      = t & 7;
    const int srcLane = (t < 8) ? t : (t + 16);
    float dp = 0.0f, dn = 0.0f;
#pragma unroll
    for (int w = 0; w < 4; ++w) {
      dp += sC[((w * 2 + 0) * 8 + vg) * 32 + srcLane];
      dn += sC[((w * 2 + 1) * 8 + vg) * 32 + srcLane];
    }
    const int i = base + t;
    const int p = pidx[i];
    const int n = nidx[i];
    const float de2 = (float)DIM * EPS * EPS;
    float ip = q[i] + q[p] - 2.0f * inv[i] * inv[p] * dp
               + 2.0f * EPS * (se[i] - se[p]) + de2;
    float in_ = q[i] + q[n] - 2.0f * inv[i] * inv[n] * dn
                + 2.0f * EPS * (se[i] - se[n]) + de2;
    ip  = fmaxf(ip, 0.0f);
    in_ = fmaxf(in_, 0.0f);
    float dpos = sqrtf(ip) + EPS;
    float dneg = sqrtf(in_) + EPS;
    float pl = dpos * dpos;
    float h  = fmaxf(MARGIN - dneg, EPS);
    float nl = h * h;
    atomicAdd(out, (pl + nl) * (1.0f / (2.0f * (float)N_SAMPLES)));
  }
}

extern "C" void kernel_launch(void* const* d_in, const int* in_sizes, int n_in,
                              void* d_out, int out_size, void* d_ws, size_t ws_size,
                              hipStream_t stream) {
  (void)in_sizes; (void)n_in; (void)out_size; (void)ws_size;
  const float* emb  = (const float*)d_in[0];
  // d_in[1] = labels (unused by the loss math)
  const int*   pidx = (const int*)d_in[2];
  const int*   nidx = (const int*)d_in[3];
  float* out = (float*)d_out;

  float* ws  = (float*)d_ws;                 // 3 * 16384 floats = 192 KB
  float* inv = ws;
  float* se  = ws + N_SAMPLES;
  float* q   = ws + 2 * N_SAMPLES;

  cl_zero_out<<<1, 1, 0, stream>>>(out);
  cl_row_stats<<<N_SAMPLES / 8, 256, 0, stream>>>(emb, inv, se, q);

  const size_t smem = (size_t)(48 * ROWP + 4 * 2 * 8 * 32) * sizeof(float);
  cl_pair_loss<<<N_SAMPLES / 16, 128, smem, stream>>>(emb, pidx, nidx,
                                                      inv, se, q, out);
}
